// MambaMIL2D_41618233099172
// MI455X (gfx1250) — matrix operand order repaired and checked by reading, not compiled, not run on hardware
//
#include <hip/hip_runtime.h>
#include <hip/hip_bf16.h>

// ---------------- problem constants ----------------
constexpr int LSEQ = 12000;
constexpr int IN   = 1024;
constexpr int DMOD = 128;
constexpr int ED   = 256;
constexpr int NST  = 16;
constexpr int DTR  = 8;
constexpr int KCONV= 4;
constexpr int NLAY = 2;
constexpr int XPAD = 48;            // xproj N=40 padded to 3 n-tiles
constexpr int MT   = LSEQ / 16;     // 750 m-tiles
constexpr int SCHUNK = 160;         // scan chunk length
constexpr int NCH  = LSEQ / SCHUNK; // 75 chunks

typedef __attribute__((ext_vector_type(16))) _Float16 v16h;
typedef __attribute__((ext_vector_type(8)))  float    v8f;

// ---------------- WMMA fragment index helpers ----------------
// A fragment (16x32 f16): lane l holds row M=l%16; element e -> K:
//   lanes 0-15 : e<8 -> K=e,     e>=8 -> K=e+8   (0..7, 16..23)
//   lanes 16-31: e<8 -> K=e+8,   e>=8 -> K=e+16  (8..15, 24..31)
__device__ __forceinline__ int packA_index(int l, int k, int KT) {
  int mt = l >> 4, r = l & 15;
  int kt = k >> 5, kk = k & 31;
  int lane = r + (((kk >> 3) & 1) << 4);
  int e = (kk & 7) + ((kk & 16) ? 8 : 0);
  return (((mt * KT + kt) << 9) + (lane << 4) + e);
}

// ---------------- weight packing: B fragment (32x16 f16) ----------------
__global__ void pack_b_kernel(const float* __restrict__ src, _Float16* __restrict__ dst,
                              int Ksrc, int Nsrc, int KT, int NT) {
  int tid = blockIdx.x * blockDim.x + threadIdx.x;
  int total = KT * NT * 512;
  if (tid >= total) return;
  int f = tid >> 9;
  int idx = tid & 511;
  int lane = idx >> 4;
  int e = idx & 15;
  int kt = f / NT, nt = f % NT;
  int k = kt * 32 + ((lane & 16) ? 8 : 0) + (e & 7) + ((e & 8) ? 16 : 0);
  int n = nt * 16 + (lane & 15);
  float v = (k < Ksrc && n < Nsrc) ? src[k * Nsrc + n] : 0.0f;
  dst[tid] = (_Float16)v;
}

// ---------------- generic packed-A x packed-B WMMA GEMM ----------------
// mode 0: store f32 ; mode 1: bias + tanh ; mode 2: residual add (out += c)
__global__ void gemm_packed(const _Float16* __restrict__ Ap, const _Float16* __restrict__ Bp,
                            float* __restrict__ out, int KT, int NT, int ldc, int mode,
                            const float* __restrict__ bias) {
  int lane = threadIdx.x & 31;
  int wave = threadIdx.x >> 5;
  int nt = blockIdx.y * (blockDim.x >> 5) + wave;
  if (nt >= NT) return;
  int mt = blockIdx.x;
  v8f c = {};
  for (int kt = 0; kt < KT; ++kt) {
    v16h a = *(const v16h*)(Ap + (((mt * KT + kt) << 9) + (lane << 4)));
    v16h b = *(const v16h*)(Bp + (((kt * NT + nt) << 9) + (lane << 4)));
    c = __builtin_amdgcn_wmma_f32_16x16x32_f16(false, a, false, b, (short)0, c, false, false);
  }
  int row0 = mt * 16 + ((lane & 16) ? 8 : 0);
  int col  = nt * 16 + (lane & 15);
  if (mode == 1) {
    float bc = bias ? bias[col] : 0.0f;
#pragma unroll
    for (int v = 0; v < 8; ++v) out[(row0 + v) * ldc + col] = tanhf(c[v] + bc);
  } else if (mode == 2) {
#pragma unroll
    for (int v = 0; v < 8; ++v) out[(row0 + v) * ldc + col] += c[v];
  } else {
#pragma unroll
    for (int v = 0; v < 8; ++v) out[(row0 + v) * ldc + col] = c[v];
  }
}

// ---------------- fc1: h = gelu(x @ W + b) ----------------
// A tile (16x32 f32, 2KB) staged via global_load_async_to_lds_b64 (ASYNCcnt path),
// converted to f16 at fragment-build time.
__global__ void gemm_fc1(const float* __restrict__ x, const _Float16* __restrict__ Bp,
                         const float* __restrict__ bias, float* __restrict__ out) {
  __shared__ __align__(16) float Asf[16 * 32];
  int lane = threadIdx.x & 31;
  int wave = threadIdx.x >> 5;           // n-tile 0..7
  int mt = blockIdx.x;
  int r  = lane & 15;
  int kb = (lane & 16) ? 8 : 0;
  int tr = threadIdx.x >> 4;             // staging row 0..15
  int tp = (threadIdx.x & 15) * 2;       // float pair within row
  unsigned lds_off = (unsigned)(uintptr_t)(&Asf[tr * 32 + tp]);
  v8f c = {};
  for (int kt = 0; kt < 32; ++kt) {
    unsigned goff = ((unsigned)(mt * 16 + tr) * IN + (unsigned)(kt * 32 + tp)) * 4u;
    __syncthreads();  // WAR: everyone done reading previous tile
    asm volatile("global_load_async_to_lds_b64 %0, %1, %2"
                 :: "v"(lds_off), "v"(goff), "s"(x)
                 : "memory");
    if (kt < 31)
      __builtin_prefetch((const void*)(x + (size_t)(mt * 16 + tr) * IN + (kt + 1) * 32 + tp), 0, 1);
    asm volatile("s_wait_asynccnt 0" ::: "memory");
    __syncthreads();  // RAW: all waves' async data landed in LDS
    float4 q0 = *(const float4*)&Asf[r * 32 + kb];
    float4 q1 = *(const float4*)&Asf[r * 32 + kb + 4];
    float4 q2 = *(const float4*)&Asf[r * 32 + kb + 16];
    float4 q3 = *(const float4*)&Asf[r * 32 + kb + 20];
    v16h a;
    a[0]=(_Float16)q0.x; a[1]=(_Float16)q0.y; a[2]=(_Float16)q0.z; a[3]=(_Float16)q0.w;
    a[4]=(_Float16)q1.x; a[5]=(_Float16)q1.y; a[6]=(_Float16)q1.z; a[7]=(_Float16)q1.w;
    a[8]=(_Float16)q2.x; a[9]=(_Float16)q2.y; a[10]=(_Float16)q2.z; a[11]=(_Float16)q2.w;
    a[12]=(_Float16)q3.x; a[13]=(_Float16)q3.y; a[14]=(_Float16)q3.z; a[15]=(_Float16)q3.w;
    v16h b = *(const v16h*)(Bp + (((kt * 8 + wave) << 9) + (lane << 4)));
    c = __builtin_amdgcn_wmma_f32_16x16x32_f16(false, a, false, b, (short)0, c, false, false);
  }
  int row0 = mt * 16 + ((lane & 16) ? 8 : 0);
  int col  = wave * 16 + (lane & 15);
  float bc = bias[col];
#pragma unroll
  for (int v = 0; v < 8; ++v) {
    float t = c[v] + bc;
    out[(row0 + v) * DMOD + col] = 0.5f * t * (1.0f + erff(t * 0.70710678118654752f));
  }
}

// ---------------- rmsnorm + pack to A-fragments (KT=4) ----------------
__global__ void rmsnorm_pack(const float* __restrict__ h, const float* __restrict__ w,
                             _Float16* __restrict__ Ap) {
  int l = blockIdx.x;
  int d = threadIdx.x;
  float v = h[l * DMOD + d];
  __shared__ float red[DMOD];
  red[d] = v * v;
  __syncthreads();
  for (int s = 64; s > 0; s >>= 1) { if (d < s) red[d] += red[d + s]; __syncthreads(); }
  float scale = rsqrtf(red[0] * (1.0f / DMOD) + 1e-5f);
  Ap[packA_index(l, d, 4)] = (_Float16)(v * scale * w[d]);
}

// ---------------- causal depthwise conv + silu + pack (KT=8) ----------------
__global__ void conv_silu_pack(const float* __restrict__ xz, const float* __restrict__ cw,
                               const float* __restrict__ cb, float* __restrict__ xc,
                               _Float16* __restrict__ Ap) {
  int idx = blockIdx.x * blockDim.x + threadIdx.x;
  if (idx >= LSEQ * ED) return;
  int l = idx >> 8, e = idx & 255;
  float acc = cb[e];
#pragma unroll
  for (int j = 0; j < KCONV; ++j) {
    int ll = l - (KCONV - 1) + j;
    if (ll >= 0) acc += xz[(size_t)ll * (2 * ED) + e] * cw[e * KCONV + j];
  }
  float s = acc / (1.0f + expf(-acc));
  xc[idx] = s;
  Ap[packA_index(l, e, 8)] = (_Float16)s;
}

// ---------------- delta = softplus(dlt @ dtw + dtb) ----------------
__global__ void delta_kernel(const float* __restrict__ dbl, const float* __restrict__ dtw,
                             const float* __restrict__ dtb, float* __restrict__ delta) {
  int idx = blockIdx.x * blockDim.x + threadIdx.x;
  if (idx >= LSEQ * ED) return;
  int l = idx >> 8, e = idx & 255;
  float acc = dtb[e];
#pragma unroll
  for (int r = 0; r < DTR; ++r) acc += dbl[l * XPAD + r] * dtw[r * ED + e];
  delta[idx] = (acc > 20.0f) ? acc : log1pf(expf(acc));
}

// ---------------- chunked parallel selective scan ----------------
// h_l = a_l*h_{l-1} + b_l is associative -> per-chunk (prodA, local state),
// sequential carry over 75 chunks, then replay with corrected initial state.
__global__ void scan_pass1(const float* __restrict__ delta, const float* __restrict__ xc,
                           const float* __restrict__ dbl, const float* __restrict__ alog,
                           float* __restrict__ chA, float* __restrict__ chB) {
  int tid = blockIdx.x * blockDim.x + threadIdx.x;  // NCH*4096 threads
  int c = tid >> 12;
  int en = tid & 4095;
  int e = en >> 4, n = en & 15;
  float Af = -expf(alog[e * NST + n]);
  float hs = 0.0f, pA = 1.0f;
  int l0 = c * SCHUNK;
  for (int i = 0; i < SCHUNK; ++i) {
    int l = l0 + i;
    float dlt = delta[l * ED + e];
    float xv  = xc[l * ED + e];
    float Bn  = dbl[l * XPAD + DTR + n];
    float a   = __expf(dlt * Af);
    hs = a * hs + dlt * xv * Bn;
    pA *= a;
  }
  chA[c * 4096 + en] = pA;
  chB[c * 4096 + en] = hs;
}

__global__ void scan_fix(const float* __restrict__ chA, const float* __restrict__ chB,
                         float* __restrict__ chInit) {
  int en = blockIdx.x * blockDim.x + threadIdx.x;   // 4096 threads
  float carry = 0.0f;
  for (int c = 0; c < NCH; ++c) {
    chInit[c * 4096 + en] = carry;
    carry = chA[c * 4096 + en] * carry + chB[c * 4096 + en];
  }
}

__global__ void scan_pass2(const float* __restrict__ delta, const float* __restrict__ xc,
                           const float* __restrict__ dbl, const float* __restrict__ alog,
                           const float* __restrict__ chInit, float* __restrict__ y) {
  int tid = blockIdx.x * blockDim.x + threadIdx.x;  // NCH*4096 threads
  int c = tid >> 12;
  int en = tid & 4095;
  int e = en >> 4, n = en & 15;                     // n == lane&15 within each 16-lane group
  float Af = -expf(alog[e * NST + n]);
  float hs = chInit[c * 4096 + en];
  int l0 = c * SCHUNK;
  for (int i = 0; i < SCHUNK; ++i) {
    int l = l0 + i;
    float dlt = delta[l * ED + e];
    float xv  = xc[l * ED + e];
    float Bn  = dbl[l * XPAD + DTR + n];
    float Cn  = dbl[l * XPAD + DTR + NST + n];
    float a   = __expf(dlt * Af);
    hs = a * hs + dlt * xv * Bn;
    float p = hs * Cn;
    p += __shfl_xor(p, 1);
    p += __shfl_xor(p, 2);
    p += __shfl_xor(p, 4);
    p += __shfl_xor(p, 8);
    if (n == 0) y[l * ED + e] = p;
  }
}

// ---------------- gating: (y + D*xc) * silu(z), pack (KT=8) ----------------
__global__ void gate_pack(const float* __restrict__ y, const float* __restrict__ xc,
                          const float* __restrict__ xz, const float* __restrict__ dp,
                          _Float16* __restrict__ Ap) {
  int idx = blockIdx.x * blockDim.x + threadIdx.x;
  if (idx >= LSEQ * ED) return;
  int l = idx >> 8, e = idx & 255;
  float z = xz[(size_t)l * (2 * ED) + ED + e];
  float g = z / (1.0f + expf(-z));
  float o = (y[idx] + dp[e] * xc[idx]) * g;
  Ap[packA_index(l, e, 8)] = (_Float16)o;
}

// ---------------- layernorm + pack (KT=4), keep f32 copy ----------------
__global__ void layernorm_pack(const float* __restrict__ h, const float* __restrict__ w,
                               const float* __restrict__ b, float* __restrict__ hln,
                               _Float16* __restrict__ Ap) {
  int l = blockIdx.x;
  int d = threadIdx.x;
  float v = h[l * DMOD + d];
  __shared__ float r1[DMOD], r2[DMOD];
  r1[d] = v; r2[d] = v * v;
  __syncthreads();
  for (int s = 64; s > 0; s >>= 1) { if (d < s) { r1[d] += r1[d + s]; r2[d] += r2[d + s]; } __syncthreads(); }
  float m   = r1[0] * (1.0f / DMOD);
  float var = r2[0] * (1.0f / DMOD) - m * m;
  float o = (v - m) * rsqrtf(var + 1e-5f) * w[d] + b[d];
  hln[l * DMOD + d] = o;
  Ap[packA_index(l, d, 4)] = (_Float16)o;
}

// ---------------- attention head ----------------
__global__ void score_kernel(const float* __restrict__ s1, const float* __restrict__ aw2,
                             const float* __restrict__ ab2, float* __restrict__ score) {
  int l = blockIdx.x * blockDim.x + threadIdx.x;
  if (l >= LSEQ) return;
  float acc = ab2[0];
  for (int d = 0; d < DMOD; ++d) acc += s1[l * DMOD + d] * aw2[d];
  score[l] = acc;
}

__global__ void softmax_reduce(const float* __restrict__ score, float* __restrict__ red) {
  __shared__ float sm[1024];
  int t = threadIdx.x;
  float m = -1e30f;
  for (int i = t; i < LSEQ; i += 1024) m = fmaxf(m, score[i]);
  sm[t] = m; __syncthreads();
  for (int s = 512; s > 0; s >>= 1) { if (t < s) sm[t] = fmaxf(sm[t], sm[t + s]); __syncthreads(); }
  float mx = sm[0]; __syncthreads();
  float sum = 0.0f;
  for (int i = t; i < LSEQ; i += 1024) sum += expf(score[i] - mx);
  sm[t] = sum; __syncthreads();
  for (int s = 512; s > 0; s >>= 1) { if (t < s) sm[t] += sm[t + s]; __syncthreads(); }
  if (t == 0) { red[0] = mx; red[1] = 1.0f / sm[0]; }
}

__global__ void zero_pooled(float* __restrict__ pooled) { pooled[threadIdx.x] = 0.0f; }

__global__ void pooled_kernel(const float* __restrict__ hln, const float* __restrict__ score,
                              const float* __restrict__ red, float* __restrict__ pooled) {
  int d = threadIdx.x;
  int l0 = blockIdx.x * 250, l1 = l0 + 250;
  float mx = red[0], inv = red[1];
  float acc = 0.0f;
  for (int l = l0; l < l1; ++l) acc += expf(score[l] - mx) * inv * hln[l * DMOD + d];
  atomicAdd(&pooled[d], acc);
}

__global__ void final_kernel(const float* __restrict__ pooled, const float* __restrict__ cw,
                             const float* __restrict__ cb, float* __restrict__ out) {
  if (threadIdx.x != 0) return;
  float l0 = cb[0], l1 = cb[1];
  for (int d = 0; d < DMOD; ++d) { l0 += pooled[d] * cw[d * 2]; l1 += pooled[d] * cw[d * 2 + 1]; }
  out[0] = l0; out[1] = l1;
  float m = fmaxf(l0, l1);
  float e0 = expf(l0 - m), e1 = expf(l1 - m), s = e0 + e1;
  out[2] = e0 / s; out[3] = e1 / s;
  out[4] = (l1 > l0) ? 1.0f : 0.0f;
}

// ---------------- host side ----------------
extern "C" void kernel_launch(void* const* d_in, const int* in_sizes, int n_in,
                              void* d_out, int out_size, void* d_ws, size_t ws_size,
                              hipStream_t stream) {
  const float* x        = (const float*)d_in[0];
  const float* fc1_w    = (const float*)d_in[2];
  const float* fc1_b    = (const float*)d_in[3];
  const float* rms_w    = (const float*)d_in[4];
  const float* inproj_w = (const float*)d_in[5];
  const float* conv_w   = (const float*)d_in[6];
  const float* conv_b   = (const float*)d_in[7];
  const float* xproj_w  = (const float*)d_in[8];
  const float* dt_w     = (const float*)d_in[9];
  const float* dt_b     = (const float*)d_in[10];
  const float* A_log    = (const float*)d_in[11];
  const float* D_p      = (const float*)d_in[12];
  const float* outproj_w= (const float*)d_in[13];
  const float* ln_w     = (const float*)d_in[14];
  const float* ln_b     = (const float*)d_in[15];
  const float* att_w1   = (const float*)d_in[16];
  const float* att_b1   = (const float*)d_in[17];
  const float* att_w2   = (const float*)d_in[18];
  const float* att_b2   = (const float*)d_in[19];
  const float* cls_w    = (const float*)d_in[20];
  const float* cls_b    = (const float*)d_in[21];

  char* ws = (char*)d_ws;
  auto carve = [&](size_t bytes) { char* p = ws; ws += (bytes + 255) & ~(size_t)255; return p; };

  float*     h      = (float*)    carve((size_t)LSEQ * DMOD * 4);
  _Float16*  hn16   = (_Float16*) carve((size_t)LSEQ * DMOD * 2);
  float*     xz     = (float*)    carve((size_t)LSEQ * 2 * ED * 4);
  float*     xc     = (float*)    carve((size_t)LSEQ * ED * 4);
  _Float16*  xc16   = (_Float16*) carve((size_t)LSEQ * ED * 2);
  float*     dbl    = (float*)    carve((size_t)LSEQ * XPAD * 4);
  float*     delta  = (float*)    carve((size_t)LSEQ * ED * 4);
  float*     yscan  = (float*)    carve((size_t)LSEQ * ED * 4);
  _Float16*  yg16   = (_Float16*) carve((size_t)LSEQ * ED * 2);
  float*     hln    = (float*)    carve((size_t)LSEQ * DMOD * 4);
  _Float16*  hln16  = (_Float16*) carve((size_t)LSEQ * DMOD * 2);
  float*     s1     = (float*)    carve((size_t)LSEQ * DMOD * 4);
  float*     score  = (float*)    carve((size_t)LSEQ * 4);
  float*     red    = (float*)    carve(256);
  float*     pooled = (float*)    carve(DMOD * 4);
  float*     chA    = (float*)    carve((size_t)NCH * 4096 * 4);
  float*     chB    = (float*)    carve((size_t)NCH * 4096 * 4);
  float*     chInit = (float*)    carve((size_t)NCH * 4096 * 4);
  _Float16*  fc1w16 = (_Float16*) carve((size_t)IN * DMOD * 2);
  _Float16*  ipw16  = (_Float16*) carve((size_t)NLAY * DMOD * 2 * ED * 2);
  _Float16*  xpw16  = (_Float16*) carve((size_t)NLAY * ED * XPAD * 2);
  _Float16*  opw16  = (_Float16*) carve((size_t)NLAY * ED * DMOD * 2);
  _Float16*  attw16 = (_Float16*) carve((size_t)DMOD * DMOD * 2);

  auto packGrid = [](int KT, int NT) { return (KT * NT * 512 + 255) / 256; };

  // ---- weight packing ----
  pack_b_kernel<<<packGrid(32, 8), 256, 0, stream>>>(fc1_w, fc1w16, IN, DMOD, 32, 8);
  for (int i = 0; i < NLAY; ++i) {
    pack_b_kernel<<<packGrid(4, 32), 256, 0, stream>>>(inproj_w + (size_t)i * DMOD * 2 * ED,
                                                       ipw16 + (size_t)i * DMOD * 2 * ED, DMOD, 2 * ED, 4, 32);
    pack_b_kernel<<<packGrid(8, 3), 256, 0, stream>>>(xproj_w + (size_t)i * ED * 40,
                                                      xpw16 + (size_t)i * ED * XPAD, ED, 40, 8, 3);
    pack_b_kernel<<<packGrid(8, 8), 256, 0, stream>>>(outproj_w + (size_t)i * ED * DMOD,
                                                      opw16 + (size_t)i * ED * DMOD, ED, DMOD, 8, 8);
  }
  pack_b_kernel<<<packGrid(4, 8), 256, 0, stream>>>(att_w1, attw16, DMOD, DMOD, 4, 8);

  // ---- stem ----
  gemm_fc1<<<MT, 256, 0, stream>>>(x, fc1w16, fc1_b, h);

  // ---- mamba layers ----
  const int EW = (LSEQ * ED + 255) / 256;
  for (int i = 0; i < NLAY; ++i) {
    rmsnorm_pack<<<LSEQ, DMOD, 0, stream>>>(h, rms_w + i * DMOD, hn16);
    gemm_packed<<<dim3(MT, 4), 256, 0, stream>>>(hn16, ipw16 + (size_t)i * DMOD * 2 * ED,
                                                 xz, 4, 32, 2 * ED, 0, nullptr);
    conv_silu_pack<<<EW, 256, 0, stream>>>(xz, conv_w + i * ED * KCONV, conv_b + i * ED, xc, xc16);
    gemm_packed<<<dim3(MT, 1), 96, 0, stream>>>(xc16, xpw16 + (size_t)i * ED * XPAD,
                                                dbl, 8, 3, XPAD, 0, nullptr);
    delta_kernel<<<EW, 256, 0, stream>>>(dbl, dt_w + i * DTR * ED, dt_b + i * ED, delta);
    const float* alog_i = A_log + (size_t)i * ED * NST;
    scan_pass1<<<NCH * 16, 256, 0, stream>>>(delta, xc, dbl, alog_i, chA, chB);
    scan_fix<<<16, 256, 0, stream>>>(chA, chB, chInit);
    scan_pass2<<<NCH * 16, 256, 0, stream>>>(delta, xc, dbl, alog_i, chInit, yscan);
    gate_pack<<<EW, 256, 0, stream>>>(yscan, xc, xz, D_p + i * ED, yg16);
    gemm_packed<<<dim3(MT, 1), 256, 0, stream>>>(yg16, opw16 + (size_t)i * ED * DMOD,
                                                 h, 8, 8, DMOD, 2, nullptr);
  }

  // ---- head ----
  layernorm_pack<<<LSEQ, DMOD, 0, stream>>>(h, ln_w, ln_b, hln, hln16);
  gemm_packed<<<dim3(MT, 1), 256, 0, stream>>>(hln16, attw16, s1, 4, 8, DMOD, 1, att_b1);
  score_kernel<<<(LSEQ + 255) / 256, 256, 0, stream>>>(s1, att_w2, att_b2, score);
  softmax_reduce<<<1, 1024, 0, stream>>>(score, red);
  zero_pooled<<<1, DMOD, 0, stream>>>(pooled);
  pooled_kernel<<<48, DMOD, 0, stream>>>(hln, score, red, pooled);
  final_kernel<<<1, 32, 0, stream>>>(pooled, cls_w, cls_b, (float*)d_out);
}
